// GATActor_83365315215888
// MI455X (gfx1250) — matrix operand order repaired and checked by reading, not compiled, not run on hardware
//
#include <hip/hip_runtime.h>
#include <hip/hip_bf16.h>

// ---------------- problem constants (match reference) ----------------
constexpr int B_   = 32;
constexpr int N_   = 512;
constexpr int F_   = 64;
constexpr int H_   = 8;
constexpr int HID_ = 64;
constexpr int OUT_ = 64;
constexpr int ACT_ = 4;
constexpr float ALPHA_ = 0.2f;

typedef __attribute__((ext_vector_type(16))) _Float16 v16h;
typedef __attribute__((ext_vector_type(8)))  float    v8f;
typedef __attribute__((ext_vector_type(4)))  unsigned int u32x4;
typedef __attribute__((ext_vector_type(8)))  int i32x8;
typedef __attribute__((ext_vector_type(4)))  int i32x4;

// ---------------- TDM (Tensor Data Mover) staging, gfx1250 ----------------
#if defined(__has_builtin)
#if __has_builtin(__builtin_amdgcn_tensor_load_to_lds) && __has_builtin(__builtin_amdgcn_s_wait_tensorcnt)
#define HAS_TDM 1
#endif
#endif
#ifndef HAS_TDM
#define HAS_TDM 0
#endif

#if HAS_TDM
// Issue a TDM DMA of `bytes` contiguous bytes global -> LDS (one wave issues).
// D# per ISA ch.8: group0 = {count=1 | lds_addr | global_addr(57) | type=2},
// group1 = {mask=0, data_size=3 (8B), tensor_dim0 = tile_dim0 = bytes/8,
//           tensor_dim1 = tile_dim1 = 1, stride0 = bytes/8}.
__device__ __forceinline__ void tdm_load_to_lds(void* ldsDst, const void* gSrc,
                                                unsigned bytes) {
  const unsigned lds_addr = (unsigned)(unsigned long long)ldsDst;  // addr[31:0] = LDS offset
  const unsigned long long ga = (unsigned long long)gSrc;
  const unsigned units = bytes >> 3;  // 8-byte elements
  u32x4 g0;
  g0[0] = 1u;                                            // count=1 (valid), user mode
  g0[1] = lds_addr;                                      // bits 63:32
  g0[2] = (unsigned)(ga & 0xffffffffu);                  // global_addr lo
  g0[3] = (unsigned)((ga >> 32) & 0x01ffffffu) | (2u << 30);  // addr hi | type=2
  i32x8 g1;
  g1[0] = (int)(3u << 16);                               // data_size=3 (8B)
  g1[1] = (int)((units & 0xffffu) << 16);                // tensor_dim0[15:0]
  g1[2] = (int)((units >> 16) & 0xffffu) | (1 << 16);    // tensor_dim0[31:16] | tensor_dim1=1
  g1[3] = (int)((units & 0xffffu) << 16);                // tile_dim0
  g1[4] = 1;                                             // tile_dim1=1, tile_dim2=0
  g1[5] = (int)units;                                    // tensor_dim0_stride lo32
  g1[6] = 0;                                             // stride0 hi | stride1 lo
  g1[7] = 0;
  i32x4 z4 = {0, 0, 0, 0};
#if __clang_major__ >= 23
  i32x8 z8 = {0, 0, 0, 0, 0, 0, 0, 0};
  __builtin_amdgcn_tensor_load_to_lds(g0, g1, z4, z4, z8, 0);
#else
  __builtin_amdgcn_tensor_load_to_lds(g0, g1, z4, z4, 0);
#endif
}
#endif  // HAS_TDM

// Fallback / generic cooperative b128 copy (16B granules).
__device__ __forceinline__ void coop_copy16(void* dst, const void* src, int bytes,
                                            int tid, int nthreads) {
  uint4* d = (uint4*)dst;
  const uint4* s = (const uint4*)src;
  for (int i = tid; i < bytes / 16; i += nthreads) d[i] = s[i];
}

// ---------------- WMMA helpers (CDNA5 gfx1250, wave32) ----------------
__device__ __forceinline__ v8f wmma16(v16h a, v16h b, v8f c) {
  return __builtin_amdgcn_wmma_f32_16x16x32_f16(
      false, a, false, b, (short)0, c, false, false);
}
__device__ __forceinline__ v8f v8f_zero() {
  v8f z = {0.f, 0.f, 0.f, 0.f, 0.f, 0.f, 0.f, 0.f};
  return z;
}

// ---------------- WMMA operand register-image swizzles ----------------
__device__ __forceinline__ int a_e_of_k(int kk) {  // kk in 0..31 -> (g<<4)|e
  int g = (kk >> 3) & 1;
  int t = kk - g * 8;
  int v = (t < 8) ? (t >> 1) : (4 + ((t - 16) >> 1));
  int e = v * 2 + (t & 1);
  return (g << 4) | e;
}
__device__ __forceinline__ size_t a_swz(int m, int k, int Kdim) {
  int tile = (m >> 4) * (Kdim >> 5) + (k >> 5);
  int eg = a_e_of_k(k & 31);
  int lane = (eg & 16) | (m & 15);
  return (size_t)tile * 512 + lane * 16 + (eg & 15);
}
__device__ __forceinline__ size_t b_swz(int k, int n, int Ncols) {
  int tile = (k >> 5) * (Ncols >> 4) + (n >> 4);
  int lane = (((k >> 3) & 1) << 4) | (n & 15);
  return (size_t)tile * 512 + lane * 16 + ((k & 7) | (((k >> 4) & 1) << 3));
}

// ---------------- staging kernels: f32 -> f16 with layout permutation ----
__global__ void k_cvt_a(const float* __restrict__ src, _Float16* __restrict__ dst,
                        int K, int total) {
  for (int i = blockIdx.x * blockDim.x + threadIdx.x; i < total;
       i += gridDim.x * blockDim.x) {
    int m = i / K, k = i - m * K;
    dst[a_swz(m, k, K)] = (_Float16)src[i];
  }
}
__global__ void k_cvt_b(const float* __restrict__ src, _Float16* __restrict__ dst,
                        int K, int Nc, int total) {
  for (int i = blockIdx.x * blockDim.x + threadIdx.x; i < total;
       i += gridDim.x * blockDim.x) {
    int bt = i / (K * Nc);
    int rem = i - bt * K * Nc;
    int k = rem / Nc, n = rem - k * Nc;
    dst[(size_t)bt * K * Nc + b_swz(k, n, Nc)] = (_Float16)src[i];
  }
}

// ---------------- kernel 1: full GAT head (per b,h) ----------------
__global__ void __launch_bounds__(256) k_gat1(
    const _Float16* __restrict__ featA,    // A-swizzled (16384 x 64)
    const _Float16* __restrict__ WheadsB,  // per-head B-swizzled (64 x 64)
    const float* __restrict__ a_src_g, const float* __restrict__ a_dst_g,
    _Float16* __restrict__ xA)             // A-swizzled (16384 x 512)
{
  extern __shared__ char smem[];
  _Float16* sWh = (_Float16*)smem;                 // 512x64 f16, B-swizzled, 64KB
  _Float16* sW  = (_Float16*)(smem + 64 * 1024);   // 64x64 f16, B-swizzled, 8KB
  float* sSrc = (float*)(smem + 64 * 1024 + 8 * 1024);
  float* sDst = sSrc + N_;
  float* sMax = sDst + N_;
  float* sInv = sMax + N_;

  const int b = blockIdx.x >> 3;
  const int h = blockIdx.x & (H_ - 1);
  const int tid = threadIdx.x;
  const int wave = tid >> 5, lane = tid & 31;
  const int r = lane & 15, g = lane >> 4;

  const float* asr = a_src_g + h * HID_;
  const float* ads = a_dst_g + h * HID_;

  // stage per-head W (already B-swizzled, 8KB) via TDM
#if HAS_TDM
  if (tid < 32) {
    tdm_load_to_lds(sW, WheadsB + (size_t)h * F_ * HID_, F_ * HID_ * 2);
    __builtin_amdgcn_s_wait_tensorcnt(0);
  }
#else
  coop_copy16(sW, WheadsB + (size_t)h * F_ * HID_, F_ * HID_ * 2, tid, blockDim.x);
#endif
  __syncthreads();

  // ---- Phase A: Wh (512x64) = feat (512x64) @ W (64x64)
  for (int it = wave; it < N_ / 16; it += 8) {
    const int row0 = it * 16;
    const size_t atile = ((size_t)b * 32 + it) * 2;
    v16h a0 = *(const v16h*)(featA + (atile + 0) * 512 + lane * 16);
    v16h a1 = *(const v16h*)(featA + (atile + 1) * 512 + lane * 16);
#pragma unroll
    for (int ot = 0; ot < HID_ / 16; ++ot) {
      v16h b0 = *(const v16h*)(sW + (0 * 4 + ot) * 512 + lane * 16);
      v16h b1 = *(const v16h*)(sW + (1 * 4 + ot) * 512 + lane * 16);
      v8f acc = v8f_zero();
      acc = wmma16(a0, b0, acc);
      acc = wmma16(a1, b1, acc);
#pragma unroll
      for (int rr = 0; rr < 8; ++rr) {
        const int row = row0 + rr + 8 * g;
        const int col = ot * 16 + r;
        sWh[b_swz(row, col, HID_)] = (_Float16)acc[rr];
      }
    }
  }
  __syncthreads();

  // ---- Phase B: s_src/s_dst = Wh . a
  for (int row = tid; row < N_; row += blockDim.x) {
    float ss = 0.f, sd = 0.f;
#pragma unroll 8
    for (int o = 0; o < HID_; ++o) {
      float w = (float)sWh[b_swz(row, o, HID_)];
      ss += w * asr[o];
      sd += w * ads[o];
    }
    sSrc[row] = ss;
    sDst[row] = sd;
  }
  __syncthreads();

  // ---- Phase C: softmax row stats over rank-1 logits
  for (int row = tid; row < N_; row += blockDim.x) {
    const float s = sSrc[row];
    float m = -1e30f;
    for (int j = 0; j < N_; ++j) {
      float e = s + sDst[j];
      e = e > 0.f ? e : ALPHA_ * e;
      m = fmaxf(m, e);
    }
    float sum = 0.f;
    for (int j = 0; j < N_; ++j) {
      float e = s + sDst[j];
      e = e > 0.f ? e : ALPHA_ * e;
      sum += __expf(e - m);
    }
    sMax[row] = m;
    sInv[row] = 1.f / sum;
  }
  __syncthreads();

  // ---- Phase D: h = elu((P @ Wh) * inv_rowsum); P built in registers
  for (int it = wave; it < N_ / 16; it += 8) {
    const int row0 = it * 16;
    const float mySrc = sSrc[row0 + r];
    const float myMax = sMax[row0 + r];
    v8f acc[4] = {v8f_zero(), v8f_zero(), v8f_zero(), v8f_zero()};
    for (int kc = 0; kc < N_ / 32; ++kc) {
      const int j0 = kc * 32;
      v8f d0 = *(const v8f*)(sDst + j0 + 8 * g);
      v8f d1 = *(const v8f*)(sDst + j0 + 16 + 8 * g);
      v16h a;
#pragma unroll
      for (int e = 0; e < 8; ++e) {
        float x = mySrc + d0[e];
        x = x > 0.f ? x : ALPHA_ * x;
        a[e] = (_Float16)__expf(x - myMax);
      }
#pragma unroll
      for (int e = 0; e < 8; ++e) {
        float x = mySrc + d1[e];
        x = x > 0.f ? x : ALPHA_ * x;
        a[e + 8] = (_Float16)__expf(x - myMax);
      }
#pragma unroll
      for (int ot = 0; ot < 4; ++ot) {
        v16h bv = *(const v16h*)(sWh + (kc * 4 + ot) * 512 + lane * 16);
        acc[ot] = wmma16(a, bv, acc[ot]);
      }
    }
#pragma unroll
    for (int ot = 0; ot < 4; ++ot)
#pragma unroll
      for (int rr = 0; rr < 8; ++rr) {
        const int row = row0 + rr + 8 * g;
        float v = acc[ot][rr] * sInv[row];
        v = v > 0.f ? v : (__expf(v) - 1.f);  // ELU
        xA[a_swz(b * N_ + row, h * HID_ + ot * 16 + r, H_ * HID_)] = (_Float16)v;
      }
  }
}

// ---------------- kernel 2: Wh2 = x (16384x512) @ W_out (512x64) ----------------
__global__ void __launch_bounds__(256) k_wh2(
    const _Float16* __restrict__ xA, const _Float16* __restrict__ WoutB,
    _Float16* __restrict__ Wh2B)
{
  extern __shared__ char smem[];
  _Float16* sB = (_Float16*)smem;  // 64KB, B-swizzled W_out
  const int tid = threadIdx.x;
  const int wave = tid >> 5, lane = tid & 31;
  const int r = lane & 15, g = lane >> 4;

#if HAS_TDM
  if (tid < 32) {
    tdm_load_to_lds(sB, WoutB, (H_ * HID_) * OUT_ * 2);
    __builtin_amdgcn_s_wait_tensorcnt(0);
  }
#else
  coop_copy16(sB, WoutB, (H_ * HID_) * OUT_ * 2, tid, blockDim.x);
#endif
  __syncthreads();

  const int tile = blockIdx.x * 8 + wave;  // 4096 tiles
  const int rowTile = tile >> 2, colTile = tile & 3;
  const int row0 = rowTile * 16, col0 = colTile * 16;
  const int KC = (H_ * HID_) / 32;  // 16

  v8f acc = v8f_zero();
  for (int kc = 0; kc < KC; ++kc) {
    v16h a  = *(const v16h*)(xA + ((size_t)rowTile * KC + kc) * 512 + lane * 16);
    v16h bv = *(const v16h*)(sB + (kc * 4 + colTile) * 512 + lane * 16);
    acc = wmma16(a, bv, acc);
  }
#pragma unroll
  for (int rr = 0; rr < 8; ++rr) {
    const int row = row0 + rr + 8 * g;
    const int col = col0 + r;
    Wh2B[(size_t)(row >> 9) * N_ * OUT_ + b_swz(row & (N_ - 1), col, OUT_)] =
        (_Float16)acc[rr];
  }
}

// ---------------- kernel 3: s2_src / s2_dst ----------------
__global__ void k_s2(const _Float16* __restrict__ Wh2B,
                     const float* __restrict__ aos, const float* __restrict__ aod,
                     float* __restrict__ s2src, float* __restrict__ s2dst)
{
  const int row = blockIdx.x * blockDim.x + threadIdx.x;
  if (row >= B_ * N_) return;
  const _Float16* p = Wh2B + (size_t)(row >> 9) * N_ * OUT_;
  const int n = row & (N_ - 1);
  float ss = 0.f, sd = 0.f;
#pragma unroll 8
  for (int o = 0; o < OUT_; ++o) {
    float w = (float)p[b_swz(n, o, OUT_)];
    ss += w * aos[o];
    sd += w * aod[o];
  }
  s2src[row] = ss;
  s2dst[row] = sd;
}

// ---------------- kernel 4: second GAT attention (per batch) ----------------
// TDM overlap: the 64KB Wh2 tile DMA is in flight while the block stages the
// score vectors and computes the softmax stats; s_wait_tensorcnt only then.
__global__ void __launch_bounds__(256) k_gat2(
    const _Float16* __restrict__ Wh2B, const float* __restrict__ s2src,
    const float* __restrict__ s2dst, _Float16* __restrict__ zA)
{
  extern __shared__ char smem[];
  _Float16* sWh = (_Float16*)smem;  // 64KB, B-swizzled Wh2_b
  float* sSrc = (float*)(smem + 64 * 1024);
  float* sDst = sSrc + N_;
  float* sMax = sDst + N_;
  float* sInv = sMax + N_;

  const int b = blockIdx.x;
  const int tid = threadIdx.x;
  const int wave = tid >> 5, lane = tid & 31;
  const int r = lane & 15, g = lane >> 4;

#if HAS_TDM
  if (tid < 32) tdm_load_to_lds(sWh, Wh2B + (size_t)b * N_ * OUT_, N_ * OUT_ * 2);
#else
  coop_copy16(sWh, Wh2B + (size_t)b * N_ * OUT_, N_ * OUT_ * 2, tid, blockDim.x);
#endif
  for (int i = tid; i < N_; i += blockDim.x) {
    sSrc[i] = s2src[b * N_ + i];
    sDst[i] = s2dst[b * N_ + i];
  }
  __syncthreads();

  // stats only touch sSrc/sDst -> overlaps with the TDM transfer
  for (int row = tid; row < N_; row += blockDim.x) {
    const float s = sSrc[row];
    float m = -1e30f;
    for (int j = 0; j < N_; ++j) {
      float e = s + sDst[j];
      e = e > 0.f ? e : ALPHA_ * e;
      m = fmaxf(m, e);
    }
    float sum = 0.f;
    for (int j = 0; j < N_; ++j) {
      float e = s + sDst[j];
      e = e > 0.f ? e : ALPHA_ * e;
      sum += __expf(e - m);
    }
    sMax[row] = m;
    sInv[row] = 1.f / sum;
  }
#if HAS_TDM
  if (tid < 32) __builtin_amdgcn_s_wait_tensorcnt(0);
#endif
  __syncthreads();

  for (int it = wave; it < N_ / 16; it += 8) {
    const int row0 = it * 16;
    const float mySrc = sSrc[row0 + r];
    const float myMax = sMax[row0 + r];
    v8f acc[4] = {v8f_zero(), v8f_zero(), v8f_zero(), v8f_zero()};
    for (int kc = 0; kc < N_ / 32; ++kc) {
      const int j0 = kc * 32;
      v8f d0 = *(const v8f*)(sDst + j0 + 8 * g);
      v8f d1 = *(const v8f*)(sDst + j0 + 16 + 8 * g);
      v16h a;
#pragma unroll
      for (int e = 0; e < 8; ++e) {
        float x = mySrc + d0[e];
        x = x > 0.f ? x : ALPHA_ * x;
        a[e] = (_Float16)__expf(x - myMax);
      }
#pragma unroll
      for (int e = 0; e < 8; ++e) {
        float x = mySrc + d1[e];
        x = x > 0.f ? x : ALPHA_ * x;
        a[e + 8] = (_Float16)__expf(x - myMax);
      }
#pragma unroll
      for (int ot = 0; ot < 4; ++ot) {
        v16h bv = *(const v16h*)(sWh + (kc * 4 + ot) * 512 + lane * 16);
        acc[ot] = wmma16(a, bv, acc[ot]);
      }
    }
#pragma unroll
    for (int ot = 0; ot < 4; ++ot)
#pragma unroll
      for (int rr = 0; rr < 8; ++rr) {
        const int row = row0 + rr + 8 * g;
        const float v = acc[ot][rr] * sInv[row];
        zA[a_swz(b, row * OUT_ + ot * 16 + r, N_ * OUT_)] = (_Float16)v;
      }
  }
}

// ---------------- kernel 5: fc1 (32x32768 @ 32768x256) + relu ----------------
__global__ void __launch_bounds__(256) k_fc1(
    const _Float16* __restrict__ zA, const _Float16* __restrict__ fc1wB,
    const float* __restrict__ fc1b, _Float16* __restrict__ z1A)
{
  __shared__ float red[8 * 256];
  const int tid = threadIdx.x;
  const int wave = tid >> 5, lane = tid & 31;
  const int r = lane & 15, g = lane >> 4;
  const int rowTile = blockIdx.x >> 4, colTile = blockIdx.x & 15;
  const int row0 = rowTile * 16, col0 = colTile * 16;
  const int KC = (N_ * OUT_) / 32;  // 1024 chunks

  v8f acc = v8f_zero();
  for (int kc = wave; kc < KC; kc += 8) {
    v16h a  = *(const v16h*)(zA + ((size_t)rowTile * KC + kc) * 512 + lane * 16);
    v16h bv = *(const v16h*)(fc1wB + ((size_t)kc * 16 + colTile) * 512 + lane * 16);
    if (kc + 8 < KC)
      __builtin_prefetch(fc1wB + ((size_t)(kc + 8) * 16 + colTile) * 512 + lane * 16, 0, 1);
    acc = wmma16(a, bv, acc);
  }
#pragma unroll
  for (int rr = 0; rr < 8; ++rr) red[wave * 256 + rr * 32 + lane] = acc[rr];
  __syncthreads();

  if (wave == 0) {
#pragma unroll
    for (int rr = 0; rr < 8; ++rr) {
      float s = 0.f;
#pragma unroll
      for (int w = 0; w < 8; ++w) s += red[w * 256 + rr * 32 + lane];
      const int row = row0 + rr + 8 * g;
      const int col = col0 + r;
      const float v = fmaxf(s + fc1b[col], 0.f);
      z1A[a_swz(row, col, 256)] = (_Float16)v;
    }
  }
}

// ---------------- kernel 6: fc2 (32x256 @ 256x256) + relu ----------------
__global__ void __launch_bounds__(256) k_fc2(
    const _Float16* __restrict__ z1A, const _Float16* __restrict__ fc2wB,
    const float* __restrict__ fc2b, _Float16* __restrict__ z2H)
{
  const int tid = threadIdx.x;
  const int wave = tid >> 5, lane = tid & 31;
  const int r = lane & 15, g = lane >> 4;
  const int tile = blockIdx.x * 8 + wave;  // 32 tiles
  const int rowTile = tile >> 4, colTile = tile & 15;
  const int row0 = rowTile * 16, col0 = colTile * 16;

  v8f acc = v8f_zero();
  for (int kc = 0; kc < 256 / 32; ++kc) {
    v16h a  = *(const v16h*)(z1A + ((size_t)rowTile * 8 + kc) * 512 + lane * 16);
    v16h bv = *(const v16h*)(fc2wB + ((size_t)kc * 16 + colTile) * 512 + lane * 16);
    acc = wmma16(a, bv, acc);
  }
#pragma unroll
  for (int rr = 0; rr < 8; ++rr) {
    const int row = row0 + rr + 8 * g;
    const int col = col0 + r;
    const float v = fmaxf(acc[rr] + fc2b[col], 0.f);
    z2H[row * 256 + col] = (_Float16)v;
  }
}

// ---------------- kernel 7: fc3 (32x256 @ 256x4) + tanh ----------------
__global__ void k_fc3(const _Float16* __restrict__ z2H, const float* __restrict__ w,
                      const float* __restrict__ bias, float* __restrict__ out)
{
  const int t = threadIdx.x;
  if (t >= B_ * ACT_) return;
  const int bb = t >> 2, j = t & 3;
  float s = bias[j];
  for (int k = 0; k < 256; ++k) s += (float)z2H[bb * 256 + k] * w[k * ACT_ + j];
  out[bb * ACT_ + j] = tanhf(s);  // MAX_ACTION == 1.0
}

// ---------------- host launcher ----------------
extern "C" void kernel_launch(void* const* d_in, const int* in_sizes, int n_in,
                              void* d_out, int out_size, void* d_ws, size_t ws_size,
                              hipStream_t stream) {
  (void)in_sizes; (void)n_in; (void)out_size; (void)ws_size;
  const float* state     = (const float*)d_in[0];
  const float* W_heads   = (const float*)d_in[1];
  const float* a_src     = (const float*)d_in[2];
  const float* a_dst     = (const float*)d_in[3];
  const float* W_out     = (const float*)d_in[4];
  const float* a_out_src = (const float*)d_in[5];
  const float* a_out_dst = (const float*)d_in[6];
  const float* fc1_w     = (const float*)d_in[7];
  const float* fc1_b     = (const float*)d_in[8];
  const float* fc2_w     = (const float*)d_in[9];
  const float* fc2_b     = (const float*)d_in[10];
  const float* fc3_w     = (const float*)d_in[11];
  const float* fc3_b     = (const float*)d_in[12];

  char* ws = (char*)d_ws;
  size_t off = 0;
  auto carve = [&](size_t bytes) {
    char* p = ws + off;
    off += (bytes + 255) & ~(size_t)255;
    return p;
  };
  _Float16* featA   = (_Float16*)carve((size_t)B_ * N_ * F_ * 2);
  _Float16* WheadsB = (_Float16*)carve((size_t)H_ * F_ * HID_ * 2);
  _Float16* WoutB   = (_Float16*)carve((size_t)H_ * HID_ * OUT_ * 2);
  _Float16* fc1wB   = (_Float16*)carve((size_t)N_ * OUT_ * 256 * 2);
  _Float16* fc2wB   = (_Float16*)carve((size_t)256 * 256 * 2);
  _Float16* xA      = (_Float16*)carve((size_t)B_ * N_ * H_ * HID_ * 2);
  _Float16* Wh2B    = (_Float16*)carve((size_t)B_ * N_ * OUT_ * 2);
  float*    s2src   = (float*)carve((size_t)B_ * N_ * 4);
  float*    s2dst   = (float*)carve((size_t)B_ * N_ * 4);
  _Float16* zA      = (_Float16*)carve((size_t)B_ * N_ * OUT_ * 2);
  _Float16* z1A     = (_Float16*)carve((size_t)B_ * 256 * 2);
  _Float16* z2H     = (_Float16*)carve((size_t)B_ * 256 * 2);

  k_cvt_a<<<1024, 256, 0, stream>>>(state, featA, F_, B_ * N_ * F_);
  k_cvt_b<<<64, 256, 0, stream>>>(W_heads, WheadsB, F_, HID_, H_ * F_ * HID_);
  k_cvt_b<<<64, 256, 0, stream>>>(W_out, WoutB, H_ * HID_, OUT_, H_ * HID_ * OUT_);
  k_cvt_b<<<2048, 256, 0, stream>>>(fc1_w, fc1wB, N_ * OUT_, 256, N_ * OUT_ * 256);
  k_cvt_b<<<64, 256, 0, stream>>>(fc2_w, fc2wB, 256, 256, 256 * 256);

  k_gat1<<<B_ * H_, 256, 80 * 1024, stream>>>(featA, WheadsB, a_src, a_dst, xA);
  k_wh2<<<512, 256, 64 * 1024, stream>>>(xA, WoutB, Wh2B);
  k_s2<<<(B_ * N_ + 255) / 256, 256, 0, stream>>>(Wh2B, a_out_src, a_out_dst, s2src, s2dst);
  k_gat2<<<B_, 256, 72 * 1024, stream>>>(Wh2B, s2src, s2dst, zA);
  k_fc1<<<32, 256, 0, stream>>>(zA, fc1wB, fc1_b, z1A);
  k_fc2<<<4, 256, 0, stream>>>(z1A, fc2wB, fc2_b, z2H);
  k_fc3<<<1, 128, 0, stream>>>(z2H, fc3_w, fc3_b, (float*)d_out);
}
